// GATEncoder_32401233281738
// MI455X (gfx1250) — compile-verified, matrix operand order checked by reading, not run on hardware
//
#include <hip/hip_runtime.h>

// ---------------------------------------------------------------------------
// GAT encoder for MI455X (gfx1250): WMMA f16 GEMM + L2-resident edge passes.
// ---------------------------------------------------------------------------

typedef _Float16 v16h __attribute__((ext_vector_type(16)));
typedef _Float16 half8 __attribute__((ext_vector_type(8)));
typedef float    v8f  __attribute__((ext_vector_type(8)));

#define IN_C   128
#define OUT_C  64
#define HEADS  2
#define HC     128          // HEADS * OUT_C
#define ALPHA  0.2f
#define BN_EPS 1e-5f

__device__ __forceinline__ float leaky(float e) { return e >= 0.f ? e : ALPHA * e; }

__device__ __forceinline__ void atomic_add_f32(float* p, float v) {
    // agent scope -> native global_atomic_add_f32 (resolves in L2)
    __hip_atomic_fetch_add(p, v, __ATOMIC_RELAXED, __HIP_MEMORY_SCOPE_AGENT);
}

__device__ __forceinline__ void atomic_max_f32(float* addr, float val) {
    if (val >= 0.f) atomicMax((int*)addr, __float_as_int(val));
    else            atomicMin((unsigned int*)addr, (unsigned int)__float_as_int(val));
}

// --- 0: zero accumulators, init segment-max to -inf-ish ---------------------
__global__ void k_init(float* __restrict__ out, float* __restrict__ m,
                       float* __restrict__ denom, float* __restrict__ sums,
                       float* __restrict__ sumsq, int n) {
    long long i = (long long)blockIdx.x * blockDim.x + threadIdx.x;
    if (i < (long long)n * HC) out[i] = 0.f;
    if (i < (long long)n * HEADS) { m[i] = -1e30f; denom[i] = 0.f; }
    if (i < HC) { sums[i] = 0.f; sumsq[i] = 0.f; }
}

// --- 1: x -> f16, W -> f16 transposed (WhT[c][k] = W[k][c]) -----------------
__global__ void k_convert(const float* __restrict__ x, const float* __restrict__ W,
                          _Float16* __restrict__ xh, _Float16* __restrict__ wht, int n) {
    long long i = (long long)blockIdx.x * blockDim.x + threadIdx.x;
    if (i < (long long)n * IN_C) xh[i] = (_Float16)x[i];
    if (i < IN_C * HC) {
        int k = (int)(i >> 7), c = (int)(i & 127);
        wht[c * IN_C + k] = (_Float16)W[k * HC + c];
    }
}

// --- 2: xp = x @ W via v_wmma_f32_16x16x32_f16 ------------------------------
// Block = 256 threads = 8 waves; wave w computes the 16x16 tile at
// rows [16*blockIdx, +16) x cols [16*w, +16). K=128 -> 4 chained WMMAs.
// 16-bit A layout: lane&15 = row, lane>>4 selects K {+8,+24} group.
__global__ __launch_bounds__(256) void k_gemm(const _Float16* __restrict__ xh,
                                              const _Float16* __restrict__ wht,
                                              float* __restrict__ xp, int n) {
    const int lane = threadIdx.x & 31;
    const int wv   = threadIdx.x >> 5;
    const int r    = lane & 15;
    const int kg   = lane >> 4;
    const int rowBase = blockIdx.x * 16;
    const int colBase = wv * 16;
    if (rowBase >= n) return;

    int arow = rowBase + r; if (arow >= n) arow = n - 1;   // clamp for ragged tail
    const _Float16* aRow = xh  + (size_t)arow * IN_C;
    const _Float16* bRow = wht + (size_t)(colBase + r) * IN_C;

    v8f acc = {};
#pragma unroll
    for (int kk = 0; kk < IN_C; kk += 32) {
        half8 a0 = *(const half8*)(aRow + kk + kg * 8);
        half8 a1 = *(const half8*)(aRow + kk + 16 + kg * 8);
        half8 b0 = *(const half8*)(bRow + kk + kg * 8);
        half8 b1 = *(const half8*)(bRow + kk + 16 + kg * 8);
        v16h A = __builtin_shufflevector(a0, a1, 0,1,2,3,4,5,6,7,8,9,10,11,12,13,14,15);
        v16h B = __builtin_shufflevector(b0, b1, 0,1,2,3,4,5,6,7,8,9,10,11,12,13,14,15);
        acc = __builtin_amdgcn_wmma_f32_16x16x32_f16(false, A, false, B,
                                                     (short)0, acc, false, false);
    }

    // C/D layout: VGPR j -> row rowBase + j (+8 for lanes>=16), col = colBase + (lane&15)
    float* dst = xp + (size_t)(rowBase + 8 * kg) * HC + colBase + r;
    if (rowBase + 16 <= n) {
        // fast path: whole tile in range -> 8 stores off one base, imm offsets
#pragma unroll
        for (int j = 0; j < 8; ++j) dst[(size_t)j * HC] = acc[j];
    } else {
#pragma unroll
        for (int j = 0; j < 8; ++j) {
            if (rowBase + 8 * kg + j < n) dst[(size_t)j * HC] = acc[j];
        }
    }
}

// --- 3: per-(node,head) attention logit halves (one wave per pair) ----------
__global__ __launch_bounds__(256) void k_logits(const float* __restrict__ xp,
                                                const float* __restrict__ att_src,
                                                const float* __restrict__ att_dst,
                                                float* __restrict__ a_src,
                                                float* __restrict__ a_dst, int n) {
    const int lane = threadIdx.x & 31;
    const int pair = blockIdx.x * 8 + (threadIdx.x >> 5);
    if (pair >= n * HEADS) return;
    const int node = pair >> 1, h = pair & 1;
    const float2 v  = *(const float2*)(xp + (size_t)node * HC + h * OUT_C + lane * 2);
    const float2 as = *(const float2*)(att_src + h * OUT_C + lane * 2);
    const float2 ad = *(const float2*)(att_dst + h * OUT_C + lane * 2);
    float ps = v.x * as.x + v.y * as.y;
    float pd = v.x * ad.x + v.y * ad.y;
#pragma unroll
    for (int off = 16; off; off >>= 1) {
        ps += __shfl_xor(ps, off, 32);
        pd += __shfl_xor(pd, off, 32);
    }
    if (lane == 0) { a_src[pair] = ps; a_dst[pair] = pd; }
}

// --- 4: segment max over destinations (self-loops are edges E..E+n-1) -------
__global__ void k_edge_max(const long long* __restrict__ ei,
                           const float* __restrict__ a_src, const float* __restrict__ a_dst,
                           float* __restrict__ m, long long E, int n) {
    long long idx = (long long)blockIdx.x * blockDim.x + threadIdx.x;
    if (idx >= E + n) return;
    int s, d;
    if (idx < E) { s = (int)ei[idx]; d = (int)ei[E + idx]; }
    else         { s = d = (int)(idx - E); }
#pragma unroll
    for (int h = 0; h < HEADS; ++h) {
        float e = leaky(a_src[s * 2 + h] + a_dst[d * 2 + h]);
        atomic_max_f32(&m[d * 2 + h], e);
    }
}

// --- 5: segment sum of exp(e - m) -------------------------------------------
__global__ void k_edge_denom(const long long* __restrict__ ei,
                             const float* __restrict__ a_src, const float* __restrict__ a_dst,
                             const float* __restrict__ m, float* __restrict__ denom,
                             long long E, int n) {
    long long idx = (long long)blockIdx.x * blockDim.x + threadIdx.x;
    if (idx >= E + n) return;
    int s, d;
    if (idx < E) { s = (int)ei[idx]; d = (int)ei[E + idx]; }
    else         { s = d = (int)(idx - E); }
#pragma unroll
    for (int h = 0; h < HEADS; ++h) {
        float e = leaky(a_src[s * 2 + h] + a_dst[d * 2 + h]);
        atomic_add_f32(&denom[d * 2 + h], __expf(e - m[d * 2 + h]));
    }
}

// --- 6: weighted scatter-aggregate; one wave per edge, 4 channels/lane ------
__global__ __launch_bounds__(256) void k_aggregate(const long long* __restrict__ ei,
                                                   const float* __restrict__ xp,
                                                   const float* __restrict__ a_src,
                                                   const float* __restrict__ a_dst,
                                                   const float* __restrict__ m,
                                                   const float* __restrict__ denom,
                                                   float* __restrict__ out,
                                                   long long E, int n) {
    const int lane = threadIdx.x & 31;
    long long edge = (long long)blockIdx.x * 8 + (threadIdx.x >> 5);
    if (edge >= E + n) return;
    int s, d;
    if (edge < E) { s = (int)ei[edge]; d = (int)ei[E + edge]; }
    else          { s = d = (int)(edge - E); }
    const int h = lane >> 4;                       // lanes 0-15: head 0, 16-31: head 1
    float e    = leaky(a_src[s * 2 + h] + a_dst[d * 2 + h]);
    float attn = __expf(e - m[d * 2 + h]) / (denom[d * 2 + h] + 1e-16f);
    const float4 v = *(const float4*)(xp + (size_t)s * HC + lane * 4);
    float* o = out + (size_t)d * HC + lane * 4;
    atomic_add_f32(o + 0, v.x * attn);
    atomic_add_f32(o + 1, v.y * attn);
    atomic_add_f32(o + 2, v.z * attn);
    atomic_add_f32(o + 3, v.w * attn);
}

// --- 7: BN batch stats (sum, sum of squares per channel) --------------------
__global__ __launch_bounds__(128) void k_bnstats(const float* __restrict__ out,
                                                 float* __restrict__ sums,
                                                 float* __restrict__ sumsq, int n) {
    const int ch = threadIdx.x;
    float s = 0.f, q = 0.f;
    for (int r = blockIdx.x; r < n; r += gridDim.x) {
        float v = out[(size_t)r * HC + ch];
        s += v; q += v * v;
    }
    atomic_add_f32(&sums[ch], s);
    atomic_add_f32(&sumsq[ch], q);
}

// --- 8: BN normalize + ReLU (bias cancels under BN; gamma/beta applied) -----
__global__ void k_bnapply(float* __restrict__ out, const float* __restrict__ sums,
                          const float* __restrict__ sumsq, const float* __restrict__ gamma,
                          const float* __restrict__ beta, int n) {
    long long i = (long long)blockIdx.x * blockDim.x + threadIdx.x;
    if (i >= (long long)n * HC) return;
    int ch = (int)(i & (HC - 1));
    float invN = 1.0f / (float)n;
    float mu  = sums[ch] * invN;
    float var = sumsq[ch] * invN - mu * mu;
    float v = (out[i] - mu) * rsqrtf(var + BN_EPS) * gamma[ch] + beta[ch];
    out[i] = v > 0.f ? v : 0.f;
}

extern "C" void kernel_launch(void* const* d_in, const int* in_sizes, int n_in,
                              void* d_out, int out_size, void* d_ws, size_t ws_size,
                              hipStream_t stream) {
    (void)n_in; (void)out_size; (void)ws_size;
    const float*     x   = (const float*)d_in[0];
    const long long* ei  = (const long long*)d_in[1];   // int64 edge_index [2,E]
    const float*     W   = (const float*)d_in[2];
    const float*     as_ = (const float*)d_in[3];
    const float*     ad_ = (const float*)d_in[4];
    // d_in[5] = bias: cancels exactly under BatchNorm, unused
    const float*     gamma = (const float*)d_in[6];
    const float*     beta  = (const float*)d_in[7];
    float* out = (float*)d_out;

    const int       N = in_sizes[0] / IN_C;
    const long long E = (long long)in_sizes[1] / 2;

    // workspace layout
    char* w = (char*)d_ws;
    float*    xp    = (float*)w;     w += (size_t)N * HC * sizeof(float);
    _Float16* xh    = (_Float16*)w;  w += (size_t)N * IN_C * sizeof(_Float16);
    _Float16* wht   = (_Float16*)w;  w += (size_t)IN_C * HC * sizeof(_Float16);
    float*    aSrc  = (float*)w;     w += (size_t)N * HEADS * sizeof(float);
    float*    aDst  = (float*)w;     w += (size_t)N * HEADS * sizeof(float);
    float*    mMax  = (float*)w;     w += (size_t)N * HEADS * sizeof(float);
    float*    denom = (float*)w;     w += (size_t)N * HEADS * sizeof(float);
    float*    sums  = (float*)w;     w += HC * sizeof(float);
    float*    sumsq = (float*)w;

    const long long totalNC = (long long)N * HC;
    const long long EN      = E + N;

    k_init<<<(unsigned)((totalNC + 255) / 256), 256, 0, stream>>>(out, mMax, denom, sums, sumsq, N);
    k_convert<<<(unsigned)(((long long)N * IN_C + 255) / 256), 256, 0, stream>>>(x, W, xh, wht, N);
    k_gemm<<<(unsigned)((N + 15) / 16), 256, 0, stream>>>(xh, wht, xp, N);
    k_logits<<<(unsigned)(((long long)N * HEADS + 7) / 8), 256, 0, stream>>>(xp, as_, ad_, aSrc, aDst, N);
    k_edge_max<<<(unsigned)((EN + 255) / 256), 256, 0, stream>>>(ei, aSrc, aDst, mMax, E, N);
    k_edge_denom<<<(unsigned)((EN + 255) / 256), 256, 0, stream>>>(ei, aSrc, aDst, mMax, denom, E, N);
    k_aggregate<<<(unsigned)((EN + 7) / 8), 256, 0, stream>>>(ei, xp, aSrc, aDst, mMax, denom, out, E, N);
    k_bnstats<<<512, 128, 0, stream>>>(out, sums, sumsq, N);
    k_bnapply<<<(unsigned)((totalNC + 255) / 256), 256, 0, stream>>>(out, sums, sumsq, gamma, beta, N);
}
